// HadamardProj_858993459241
// MI455X (gfx1250) — compile-verified, matrix operand order checked by reading, not run on hardware
//
#include <hip/hip_runtime.h>

typedef _Float16 half_t;
typedef _Float16 v16h __attribute__((ext_vector_type(16)));
typedef float v8f  __attribute__((ext_vector_type(8)));

#define IN_K   2048
#define OUT_N  10000
#define BM     128
#define BN     256
#define N_PAD  10240    // 40 * 256
#define M_ROWS 16384
#define LDSS   40       // LDS row stride in halves (80B) -> conflict-free b128 access

#if __has_builtin(__builtin_amdgcn_global_load_async_to_lds_b128)
#define USE_ASYNC 1
typedef int v4i __attribute__((vector_size(16)));
typedef __attribute__((address_space(1))) v4i gv4i;   // global-AS int4
typedef __attribute__((address_space(3))) v4i lv4i;   // LDS-AS int4
#else
#define USE_ASYNC 0
#endif

static __device__ __forceinline__ void wait_async_zero() {
#if __has_builtin(__builtin_amdgcn_s_wait_asynccnt)
  __builtin_amdgcn_s_wait_asynccnt(0);
#else
  asm volatile("s_wait_asynccnt 0" ::: "memory");
#endif
}

// ---------------- Kernel 1: L2-normalize rows of x, emit f16 ----------------
__global__ __launch_bounds__(256) void normalize_rows(const float* __restrict__ x,
                                                      half_t* __restrict__ xn) {
  __shared__ float wsum[8];
  const int row = blockIdx.x;
  const float* xr = x + (size_t)row * IN_K;

  float ss = 0.f;
#pragma unroll
  for (int i = 0; i < 8; ++i) {
    float v = xr[threadIdx.x + i * 256];
    ss += v * v;
  }
#pragma unroll
  for (int off = 16; off > 0; off >>= 1)
    ss += __shfl_xor(ss, off, 32);

  const int wid  = threadIdx.x >> 5;
  const int lane = threadIdx.x & 31;
  if (lane == 0) wsum[wid] = ss;
  __syncthreads();

  float tot = 0.f;
#pragma unroll
  for (int w = 0; w < 8; ++w) tot += wsum[w];
  const float inv = 1.0f / (sqrtf(tot) + 1e-8f);

  half_t* xo = xn + (size_t)row * IN_K;
#pragma unroll
  for (int i = 0; i < 8; ++i) {
    const int idx = threadIdx.x + i * 256;
    xo[idx] = (half_t)(xr[idx] * inv);
  }
}

// ---------------- Kernel 2: proj f32 -> f16, zero-pad rows to N_PAD ----------------
__global__ __launch_bounds__(256) void convert_proj(const float* __restrict__ proj,
                                                    half_t* __restrict__ pf) {
  const size_t i   = (size_t)blockIdx.x * 256 + threadIdx.x;  // over N_PAD*IN_K
  const size_t row = i >> 11;                                 // /IN_K
  pf[i] = (row < OUT_N) ? (half_t)proj[i] : (half_t)0.f;
}

// ---------------- Kernel 3: WMMA GEMM: out = -scale * xn @ pf^T + bias ----------------
// A = xn [M][K] f16 row-major; B = pf [N_PAD][K] f16 (K-contiguous == B columns)
__global__ __launch_bounds__(256) void had_gemm(const half_t* __restrict__ A,
                                                const half_t* __restrict__ B,
                                                const float* __restrict__ scale,
                                                const float* __restrict__ bias,
                                                float* __restrict__ out) {
  __shared__ half_t sA[2][BM * LDSS];   // 2 x 10 KB
  __shared__ half_t sB[2][BN * LDSS];   // 2 x 20 KB

  const int m0   = blockIdx.y * BM;
  const int n0   = blockIdx.x * BN;
  const int tid  = threadIdx.x;
  const int lane = tid & 31;
  const int wid  = tid >> 5;

  // 8 waves -> 2 (M) x 4 (N); each wave owns a 64x64 output sub-tile
  const int waveM = (wid >> 2) * 64;
  const int waveN = (wid & 3) * 64;

  // staging: rows x 32 halves = rows x 4 chunks(16B)
  const int ldRow   = tid >> 2;          // 0..63 base row
  const int ldChunk = (tid & 3) * 8;     // chunk offset in halves (16B granules)

  // fragment addressing per documented 16-bit A/B layout
  const int fragRow = lane & 15;
  const int kHalf   = (lane >> 4) * 8;   // lanes 0-15: K 0-7/16-23; 16-31: K 8-15/24-31

  v8f acc[4][4] = {};
  union Frag { v16h v; uint4 q[2]; };

  auto stage = [&](int buf, int kt) {
#pragma unroll
    for (int j = 0; j < 2; ++j) {        // A: 128 rows
      const int r = ldRow + j * 64;
      const half_t* g = A + (size_t)(m0 + r) * IN_K + kt + ldChunk;
      half_t* l = &sA[buf][r * LDSS + ldChunk];
#if USE_ASYNC
      __builtin_amdgcn_global_load_async_to_lds_b128((gv4i*)g, (lv4i*)l, 0, 0);
#else
      *(uint4*)l = *(const uint4*)g;
#endif
    }
#pragma unroll
    for (int j = 0; j < 4; ++j) {        // B: 256 rows
      const int r = ldRow + j * 64;
      const half_t* g = B + (size_t)(n0 + r) * IN_K + kt + ldChunk;
      half_t* l = &sB[buf][r * LDSS + ldChunk];
#if USE_ASYNC
      __builtin_amdgcn_global_load_async_to_lds_b128((gv4i*)g, (lv4i*)l, 0, 0);
#else
      *(uint4*)l = *(const uint4*)g;
#endif
    }
  };

  auto compute = [&](int buf) {
    Frag bf[4];
#pragma unroll
    for (int ni = 0; ni < 4; ++ni) {
      const half_t* pb = &sB[buf][(waveN + ni * 16 + fragRow) * LDSS + kHalf];
      bf[ni].q[0] = *(const uint4*)pb;         // K = kHalf..kHalf+7
      bf[ni].q[1] = *(const uint4*)(pb + 16);  // K = 16+kHalf..
    }
#pragma unroll
    for (int mi = 0; mi < 4; ++mi) {
      Frag af;
      const half_t* pa = &sA[buf][(waveM + mi * 16 + fragRow) * LDSS + kHalf];
      af.q[0] = *(const uint4*)pa;
      af.q[1] = *(const uint4*)(pa + 16);
#pragma unroll
      for (int ni = 0; ni < 4; ++ni) {
        acc[mi][ni] = __builtin_amdgcn_wmma_f32_16x16x32_f16(
            false, af.v, false, bf[ni].v, (short)0, acc[mi][ni], false, false);
      }
    }
  };

  // ---- double-buffered main loop ----
  stage(0, 0);
#if USE_ASYNC
  wait_async_zero();
#endif
  __syncthreads();

  int cur = 0;
  for (int kt = 0; kt < IN_K; kt += 32) {
    if (kt + 32 < IN_K) stage(cur ^ 1, kt + 32);
    compute(cur);
#if USE_ASYNC
    wait_async_zero();
#endif
    __syncthreads();
    cur ^= 1;
  }

  // epilogue: C/D layout — lane&15 -> N, lane>>4 selects M +0/+8, VGPR r -> M row r
  const float s  = -scale[0];
  const int  nc  = lane & 15;
  const int  mh  = (lane >> 4) * 8;
#pragma unroll
  for (int mi = 0; mi < 4; ++mi) {
#pragma unroll
    for (int ni = 0; ni < 4; ++ni) {
      const int n = n0 + waveN + ni * 16 + nc;
      if (n < OUT_N) {
        const float bv = bias[n];
#pragma unroll
        for (int r = 0; r < 8; ++r) {
          const int m = m0 + waveM + mi * 16 + mh + r;
          out[(size_t)m * OUT_N + n] = s * acc[mi][ni][r] + bv;
        }
      }
    }
  }
}

extern "C" void kernel_launch(void* const* d_in, const int* in_sizes, int n_in,
                              void* d_out, int out_size, void* d_ws, size_t ws_size,
                              hipStream_t stream) {
  const float* x     = (const float*)d_in[0];   // [16384, 2048]
  const float* proj  = (const float*)d_in[1];   // [10000, 2048]
  const float* scale = (const float*)d_in[2];   // [1]
  const float* bias  = (const float*)d_in[3];   // [10000]
  float* out = (float*)d_out;                   // [16384, 10000]

  half_t* xn = (half_t*)d_ws;                                        // 64 MiB
  half_t* pf = (half_t*)((char*)d_ws + (size_t)M_ROWS * IN_K * 2);   // 40 MiB

  normalize_rows<<<M_ROWS, 256, 0, stream>>>(x, xn);

  const int cvt_blocks = (int)(((size_t)N_PAD * IN_K) / 256);        // exact multiple
  convert_proj<<<cvt_blocks, 256, 0, stream>>>(proj, pf);

  dim3 grid(N_PAD / BN, M_ROWS / BM);  // 40 x 128
  had_gemm<<<grid, 256, 0, stream>>>(xn, pf, scale, bias, out);
}
